// TraceLayer_66142496358887
// MI455X (gfx1250) — matrix operand order repaired and checked by reading, not compiled, not run on hardware
//
#include <hip/hip_runtime.h>

// GEMM: out[b, po] = sum_k x[b,k] * W[po,k] + bias[po]
//   M = PO = 4096, N = B = 32, K = 16384, all fp32 in HBM.
// HBM-bound on W (256 MB, ~11 us at 23.3 TB/s). bf16 WMMA with f32 accumulate.
// Main path: ksplit=32, per-workgroup x k-chunk staged in LDS as bf16 (shared
// by the 8 waves), deterministic two-phase K reduction through d_ws.

typedef __attribute__((ext_vector_type(16))) __bf16 v16bf;
typedef __attribute__((ext_vector_type(8)))  __bf16 v8bf;
typedef __attribute__((ext_vector_type(8)))  float  v8f;
typedef __attribute__((ext_vector_type(4)))  float  v4f;

#define K_DIM  16384
#define PO_DIM 4096
#define B_DIM  32
#define NTILES (PO_DIM / 16)

// ---------------- LDS-staged main kernel: ksplit = 32, kchunk = 512 ----------
#define KCH     512
#define KSPLIT  32
#define ROWSTR  (KCH + 8)          // bf16 elements per LDS row (16 B pad -> 260-dword stride)

__global__ __launch_bounds__(256) void trace_gemm_wmma_lds(
    const float* __restrict__ W, const float* __restrict__ X,
    float* __restrict__ out)
{
  extern __shared__ char lds_raw[];
  __bf16* ldsx = (__bf16*)lds_raw;         // [32][ROWSTR] bf16: x[b, kbeg..kbeg+512)

  const int warp = (int)(threadIdx.x >> 5);
  const int lane = (int)(threadIdx.x & 31);
  const int ks   = (int)(blockIdx.x >> 5);              // 0..31 (8 waves share ks)
  const int tile = (int)((blockIdx.x & 31) * 8 + warp); // 0..255
  const int po0  = tile * 16;
  const int kbeg = ks * KCH;

  // ---- cooperative fill: convert x[0..31, kbeg..kbeg+512) fp32 -> bf16 in LDS
  #pragma unroll
  for (int it = 0; it < 8; ++it) {
    const int u  = (int)threadIdx.x + it * 256;  // 8-float unit id, 2048 total
    const int b  = u >> 6;                       // 64 units per row
    const int ko = (u & 63) * 8;
    const float* src = X + (size_t)b * K_DIM + kbeg + ko;
    v4f f0 = *(const v4f*)src;
    v4f f1 = *(const v4f*)(src + 4);
    v8bf t;
    #pragma unroll
    for (int e = 0; e < 4; ++e) {
      t[e]     = (__bf16)f0[e];
      t[e + 4] = (__bf16)f1[e];
    }
    *(v8bf*)(ldsx + b * ROWSTR + ko) = t;        // 16 B ds_store
  }
  __syncthreads();

  const int m      = lane & 15;   // A row / B column role
  const int hi     = lane >> 4;
  const int khalfA = hi * 16;     // A lane covers k in [khalfA,+16)
  const int ksecB  = hi * 16;     // B lane covers k in [ksecB,+16)

  const float*  wrow = W + (size_t)(po0 + m) * K_DIM;
  const __bf16* lb0  = ldsx + m * ROWSTR + ksecB;          // b = m
  const __bf16* lb1  = ldsx + (m + 16) * ROWSTR + ksecB;   // b = m+16

  v8f c0 = {};   // D tile b = 0..15
  v8f c1 = {};   // D tile b = 16..31

  for (int kk = 0; kk < KCH; kk += 32) {
    // ---- A tile: W[po0..po0+15, k0..k0+31] fp32 -> bf16 (16-bit A 16x32 layout)
    const float* wa = wrow + kbeg + kk + khalfA;
    __builtin_prefetch(wa + 256, 0, 0);          // stream W ahead
    v4f w0 = *(const v4f*)(wa);
    v4f w1 = *(const v4f*)(wa + 4);
    v4f w2 = *(const v4f*)(wa + 8);
    v4f w3 = *(const v4f*)(wa + 12);
    v16bf a;
    #pragma unroll
    for (int e = 0; e < 4; ++e) {
      a[e]      = (__bf16)w0[e];
      a[e + 4]  = (__bf16)w1[e];
      a[e + 8]  = (__bf16)w2[e];
      a[e + 12] = (__bf16)w3[e];
    }

    // ---- B tiles straight from LDS (already bf16, already in VGPR layout)
    v16bf b0 = *(const v16bf*)(lb0 + kk);        // 2x ds_load_b128
    v16bf b1 = *(const v16bf*)(lb1 + kk);

    c0 = __builtin_amdgcn_wmma_f32_16x16x32_bf16(false, a, false, b0, (short)0, c0, false, false);
    c1 = __builtin_amdgcn_wmma_f32_16x16x32_bf16(false, a, false, b1, (short)0, c1, false, false);
  }

  // ---- partial store: ws[(ks*32 + b) * 4096 + po]
  const int po_lo = hi * 8;
  float* o0 = out + ((size_t)(ks * B_DIM + m) * PO_DIM) + po0 + po_lo;
  float* o1 = out + ((size_t)(ks * B_DIM + 16 + m) * PO_DIM) + po0 + po_lo;
  v4f s;
  s = (v4f){c0[0], c0[1], c0[2], c0[3]}; *(v4f*)(o0)     = s;
  s = (v4f){c0[4], c0[5], c0[6], c0[7]}; *(v4f*)(o0 + 4) = s;
  s = (v4f){c1[0], c1[1], c1[2], c1[3]}; *(v4f*)(o1)     = s;
  s = (v4f){c1[4], c1[5], c1[6], c1[7]}; *(v4f*)(o1 + 4) = s;
}

// ---------------- generic fallback (small ws): global B loads ----------------
__global__ __launch_bounds__(256) void trace_gemm_wmma(
    const float* __restrict__ W, const float* __restrict__ X,
    const float* __restrict__ bias, float* __restrict__ out,
    int kchunk, int direct)
{
  const int wave = (int)((blockIdx.x * blockDim.x + threadIdx.x) >> 5);
  const int lane = (int)(threadIdx.x & 31);

  const int tile = wave % NTILES;
  const int ks   = wave / NTILES;
  const int po0  = tile * 16;
  const int kbeg = ks * kchunk;
  const int kend = kbeg + kchunk;

  const int m      = lane & 15;
  const int hi     = lane >> 4;
  const int khalfA = hi * 16;
  const int ksecB  = hi * 16;

  const float* wrow  = W + (size_t)(po0 + m) * K_DIM;
  const float* xrow0 = X + (size_t)m * K_DIM;
  const float* xrow1 = X + (size_t)(m + 16) * K_DIM;

  v8f c0 = {}; v8f c1 = {};

  for (int k0 = kbeg; k0 < kend; k0 += 32) {
    const float* wa = wrow + k0 + khalfA;
    __builtin_prefetch(wa + 256, 0, 0);
    v4f w0 = *(const v4f*)(wa);
    v4f w1 = *(const v4f*)(wa + 4);
    v4f w2 = *(const v4f*)(wa + 8);
    v4f w3 = *(const v4f*)(wa + 12);

    const float* xb0 = xrow0 + k0 + ksecB;
    v4f p0 = *(const v4f*)(xb0);
    v4f p1 = *(const v4f*)(xb0 + 4);
    v4f p2 = *(const v4f*)(xb0 + 8);
    v4f p3 = *(const v4f*)(xb0 + 12);

    const float* xb1 = xrow1 + k0 + ksecB;
    v4f q0 = *(const v4f*)(xb1);
    v4f q1 = *(const v4f*)(xb1 + 4);
    v4f q2 = *(const v4f*)(xb1 + 8);
    v4f q3 = *(const v4f*)(xb1 + 12);

    v16bf a, b0, b1;
    #pragma unroll
    for (int e = 0; e < 4; ++e) {
      a[e]       = (__bf16)w0[e];
      a[e + 4]   = (__bf16)w1[e];
      a[e + 8]   = (__bf16)w2[e];
      a[e + 12]  = (__bf16)w3[e];
      b0[e]      = (__bf16)p0[e];
      b0[e + 4]  = (__bf16)p1[e];
      b0[e + 8]  = (__bf16)p2[e];
      b0[e + 12] = (__bf16)p3[e];
      b1[e]      = (__bf16)q0[e];
      b1[e + 4]  = (__bf16)q1[e];
      b1[e + 8]  = (__bf16)q2[e];
      b1[e + 12] = (__bf16)q3[e];
    }

    c0 = __builtin_amdgcn_wmma_f32_16x16x32_bf16(false, a, false, b0, (short)0, c0, false, false);
    c1 = __builtin_amdgcn_wmma_f32_16x16x32_bf16(false, a, false, b1, (short)0, c1, false, false);
  }

  const int po_lo = hi * 8;
  if (direct) {
    float* o0 = out + (size_t)m * PO_DIM + po0 + po_lo;
    float* o1 = out + (size_t)(m + 16) * PO_DIM + po0 + po_lo;
    const float* bp = bias + po0 + po_lo;
    #pragma unroll
    for (int v = 0; v < 8; ++v) {
      o0[v] = c0[v] + bp[v];
      o1[v] = c1[v] + bp[v];
    }
  } else {
    float* o0 = out + ((size_t)(ks * B_DIM + m) * PO_DIM) + po0 + po_lo;
    float* o1 = out + ((size_t)(ks * B_DIM + 16 + m) * PO_DIM) + po0 + po_lo;
    v4f s;
    s = (v4f){c0[0], c0[1], c0[2], c0[3]}; *(v4f*)(o0)     = s;
    s = (v4f){c0[4], c0[5], c0[6], c0[7]}; *(v4f*)(o0 + 4) = s;
    s = (v4f){c1[0], c1[1], c1[2], c1[3]}; *(v4f*)(o1)     = s;
    s = (v4f){c1[4], c1[5], c1[6], c1[7]}; *(v4f*)(o1 + 4) = s;
  }
}

__global__ __launch_bounds__(256) void reduce_bias_kernel(
    const float* __restrict__ ws, const float* __restrict__ bias,
    float* __restrict__ out, int ksplit)
{
  const int idx = (int)(blockIdx.x * blockDim.x + threadIdx.x);  // 0 .. 32*4096-1
  if (idx >= B_DIM * PO_DIM) return;
  const int po = idx & (PO_DIM - 1);
  const int b  = idx >> 12;
  float s = bias[po];
  for (int k = 0; k < ksplit; ++k)
    s += ws[(size_t)(k * B_DIM + b) * PO_DIM + po];
  out[idx] = s;
}

extern "C" void kernel_launch(void* const* d_in, const int* in_sizes, int n_in,
                              void* d_out, int out_size, void* d_ws, size_t ws_size,
                              hipStream_t stream) {
  const float* x    = (const float*)d_in[0];   // [32, 128, 128]
  const float* W    = (const float*)d_in[1];   // [64, 64, 128, 128]
  const float* bias = (const float*)d_in[2];   // [64, 64]
  float* out = (float*)d_out;                  // [32, 64, 64]
  float* ws  = (float*)d_ws;

  const int n = B_DIM * PO_DIM;
  const size_t need32 = (size_t)KSPLIT * B_DIM * PO_DIM * sizeof(float);  // 16 MB

  if (need32 <= ws_size) {
    // main path: LDS-staged bf16 x, 32-way K split, 1024 workgroups x 8 waves
    const int blocks = (NTILES * KSPLIT) / 8;           // 1024
    const size_t lds_bytes = (size_t)B_DIM * ROWSTR * sizeof(__bf16);  // ~33 KB
    trace_gemm_wmma_lds<<<blocks, 256, lds_bytes, stream>>>(W, x, ws);
    reduce_bias_kernel<<<(n + 255) / 256, 256, 0, stream>>>(ws, bias, out, KSPLIT);
    return;
  }

  // fallback: pick largest K split whose partial buffer fits in ws
  int ksplit = 1;
  const int cands[4] = {16, 8, 4, 2};
  for (int i = 0; i < 4; ++i) {
    size_t need = (size_t)cands[i] * B_DIM * PO_DIM * sizeof(float);
    if (need <= ws_size) { ksplit = cands[i]; break; }
  }

  if (ksplit > 1) {
    const int waves  = NTILES * ksplit;
    const int blocks = (waves * 32) / 256;
    trace_gemm_wmma<<<blocks, 256, 0, stream>>>(W, x, bias, ws, K_DIM / ksplit, 0);
    reduce_bias_kernel<<<(n + 255) / 256, 256, 0, stream>>>(ws, bias, out, ksplit);
  } else {
    trace_gemm_wmma<<<(NTILES * 32) / 256, 256, 0, stream>>>(W, x, bias, out, K_DIM, 1);
  }
}
